// MotionTransformer_28698971472407
// MI455X (gfx1250) — matrix operand order repaired and checked by reading, not compiled, not run on hardware
//
#include <hip/hip_runtime.h>
#include <hip/hip_bf16.h>

typedef __attribute__((ext_vector_type(16))) __bf16 v16bf;
typedef __attribute__((ext_vector_type(8)))  __bf16 v8bf;
typedef __attribute__((ext_vector_type(8)))  float  v8f;

#define BB  64
#define SS  1024
#define FFK 136
#define DD  64
#define HHN 2
#define DH  32
#define DFF 256
#define MM  (BB*SS)   // 65536
#define LDP 1032      // padded LDS row (bf16 elems) for the attn stripe
#define VROW 144      // padded LDS row (bf16 elems) for the V slice (288B)

union V16U { v16bf v; v8bf h[2]; };

// ---- fragment loaders (CDNA5 wave32 WMMA layouts, ISA 7.12.2) ----

__device__ __forceinline__ v16bf load_a_bf16(const __bf16* A, int lda, int row0, int k0, int lane) {
    int hf = lane >> 4, m = lane & 15;
    const __bf16* p = A + (size_t)(row0 + m) * lda + k0 + hf * 8;
    V16U r;
    r.h[0] = *(const v8bf*)(p);        // K = k0 + 8*hf      .. +7
    r.h[1] = *(const v8bf*)(p + 16);   // K = k0 + 16 + 8*hf .. +7
    return r.v;
}

__device__ __forceinline__ v16bf load_a_f32(const float* A, int lda, int row0, int k0, int Ktot, int lane) {
    int hf = lane >> 4, m = lane & 15;
    const float* p = A + (size_t)(row0 + m) * lda + k0 + hf * 8;
    int ks0 = k0 + hf * 8, ks1 = ks0 + 16;
    V16U r;
    if (ks0 + 8 <= Ktot) {
        v8f f = *(const v8f*)(p);
        #pragma unroll
        for (int i = 0; i < 8; ++i) r.h[0][i] = (__bf16)f[i];
    } else {
        #pragma unroll
        for (int i = 0; i < 8; ++i) r.h[0][i] = (__bf16)0.0f;
    }
    if (ks1 + 8 <= Ktot) {
        v8f f = *(const v8f*)(p + 16);
        #pragma unroll
        for (int i = 0; i < 8; ++i) r.h[1][i] = (__bf16)f[i];
    } else {
        #pragma unroll
        for (int i = 0; i < 8; ++i) r.h[1][i] = (__bf16)0.0f;
    }
    return r.v;
}

__device__ __forceinline__ v16bf load_b(const __bf16* Bt, int ldb, int n0, int k0, int lane) {
    const __bf16* p = Bt + (size_t)(n0 + (lane & 15)) * ldb + k0 + ((lane >> 4) << 3);
    V16U r;
    r.h[0] = *(const v8bf*)(p);
    r.h[1] = *(const v8bf*)(p + 16);
    return r.v;
}

#define WMMA_BF16(a,b,c) __builtin_amdgcn_wmma_f32_16x16x32_bf16(false,(a),false,(b),(short)0,(c),false,false)

// ---- weight transpose + f32->bf16 convert (zero-padded K) ----
__global__ void k_tcvt(const float* __restrict__ src, __bf16* __restrict__ dst, int K, int N, int ldk) {
    int idx = blockIdx.x * blockDim.x + threadIdx.x;
    if (idx >= N * ldk) return;
    int n = idx / ldk, k = idx % ldk;
    float v = (k < K) ? src[(size_t)k * N + n] : 0.0f;
    dst[(size_t)n * ldk + k] = (__bf16)v;
}

// ---- h = x @ W_in + b_in + PE ; also bf16 copy. One wave = 16x64 tile row ----
__global__ void __launch_bounds__(256) k_in_proj(const float* __restrict__ x, const __bf16* __restrict__ Wt,
                          const float* __restrict__ bias, float* __restrict__ h, __bf16* __restrict__ hb) {
    int wave = threadIdx.x >> 5, lane = threadIdx.x & 31;
    int tm = blockIdx.x * 8 + wave;
    v8f acc[4] = {};
    for (int k0 = 0; k0 < 160; k0 += 32) {
        v16bf a = load_a_f32(x, FFK, tm * 16, k0, FFK, lane);
        #pragma unroll
        for (int nt = 0; nt < 4; ++nt)
            acc[nt] = WMMA_BF16(a, load_b(Wt, 160, nt * 16, k0, lane), acc[nt]);
    }
    int rbase = tm * 16 + ((lane >> 4) << 3);
    #pragma unroll
    for (int nt = 0; nt < 4; ++nt) {
        int n = nt * 16 + (lane & 15);
        float bn = bias[n];
        float freq = expf(-0.14391157f * (float)(n & ~1));   // ln(10000)/64
        #pragma unroll
        for (int r = 0; r < 8; ++r) {
            int row = rbase + r;
            float ang = (float)(row & (SS - 1)) * freq;
            float pe = (n & 1) ? cosf(ang) : sinf(ang);
            float v = acc[nt][r] + bn + pe;
            size_t i = (size_t)row * DD + n;
            h[i] = v; hb[i] = (__bf16)v;
        }
    }
}

// ---- qkv = h @ W_qkv + b_qkv (bf16 out); q part pre-scaled by 1/sqrt(DH) ----
__global__ void __launch_bounds__(256) k_qkv(const __bf16* __restrict__ hb, const __bf16* __restrict__ Wt,
                      const float* __restrict__ bias, __bf16* __restrict__ qkv) {
    int wave = threadIdx.x >> 5, lane = threadIdx.x & 31;
    int t = blockIdx.x * 8 + wave;
    int tm = t / 3, g = t % 3;              // group of 4 n-tiles (64 cols)
    int nbase = g * 64;
    v8f acc[4] = {};
    for (int k0 = 0; k0 < DD; k0 += 32) {
        v16bf a = load_a_bf16(hb, DD, tm * 16, k0, lane);
        #pragma unroll
        for (int nt = 0; nt < 4; ++nt)
            acc[nt] = WMMA_BF16(a, load_b(Wt, DD, nbase + nt * 16, k0, lane), acc[nt]);
    }
    int rbase = tm * 16 + ((lane >> 4) << 3);
    float scale = (g == 0) ? 0.17677669529663687f : 1.0f;  // fold 1/sqrt(DH) into q
    #pragma unroll
    for (int nt = 0; nt < 4; ++nt) {
        int n = nbase + nt * 16 + (lane & 15);
        float bn = bias[n];
        #pragma unroll
        for (int r = 0; r < 8; ++r)
            qkv[(size_t)(rbase + r) * 192 + n] = (__bf16)((acc[nt][r] + bn) * scale);
    }
}

// ---- vt[b,h,d,j] = v part of qkv (K-contiguous B operand for attn@V) ----
__global__ void k_vt(const __bf16* __restrict__ qkv, __bf16* __restrict__ vt) {
    size_t idx = (size_t)blockIdx.x * blockDim.x + threadIdx.x;
    if (idx >= (size_t)BB * HHN * DH * SS) return;
    int j  = idx & (SS - 1);
    int d  = (idx >> 10) & (DH - 1);
    int hh = (idx >> 15) & 1;
    int b  = idx >> 16;
    vt[idx] = qkv[((size_t)b * SS + j) * 192 + 2 * DD + hh * DH + d];
}

// ---- fully fused attention: scores -> softmax -> attn write (NT) -> ctx ----
// grid (64 row-tiles, 128 bh), block 256 (8 waves). Wave w owns cols [w*128, w*128+128).
__global__ void __launch_bounds__(256) k_attn_fused(const __bf16* __restrict__ qkv,
                                                    const __bf16* __restrict__ vt,
                                                    float* __restrict__ attn,
                                                    float* __restrict__ ctx) {
    __shared__ __align__(32) __bf16 Psh[16 * LDP];       // normalized probs (bf16) 16x1024
    __shared__ __align__(32) __bf16 Vsh[8 * 32 * VROW];  // per-wave V slices, async-staged
    __shared__ float red[8 * 16];                        // per-wave row stats
    __shared__ float ctile[16 * 32];                     // ctx output tile

    int wave = threadIdx.x >> 5, lane = threadIdx.x & 31;
    int hf = lane >> 4, ln = lane & 15;
    int i0 = blockIdx.x * 16;
    int bh = blockIdx.y, b = bh >> 1, hh = bh & 1;
    int j0 = wave * 128;

    // Phase 0: async-stage this wave's V slice (32 rows x 128 cols bf16) to LDS.
    // Overlaps with score/softmax compute; tracked by ASYNCcnt.
    {
        const __bf16* vsrc = vt + (size_t)bh * DH * SS + j0;   // global row stride SS
        __bf16* vdst = Vsh + wave * 32 * VROW;
        #pragma unroll
        for (int it = 0; it < 16; ++it) {
            int c = it * 32 + lane;                 // 512 chunks of 16B
            int row = c >> 4, ce = (c & 15) * 8;    // 8 bf16 per chunk
            unsigned lds = (unsigned)(size_t)(vdst + row * VROW + ce);
            unsigned long long ga = (unsigned long long)(size_t)(vsrc + (size_t)row * SS + ce);
            asm volatile("global_load_async_to_lds_b128 %0, %1, off" :: "v"(lds), "v"(ga) : "memory");
        }
    }

    const __bf16* qb = qkv + (size_t)b * SS * 192 + hh * DH;        // q (pre-scaled), lda 192
    const __bf16* kb = qkv + (size_t)b * SS * 192 + DD + hh * DH;   // k, ldb 192

    // Phase 1: scores 16 x 128 in registers (8 accumulators)
    v16bf a = load_a_bf16(qb, 192, i0, 0, lane);
    v8f acc[8];
    #pragma unroll
    for (int jt = 0; jt < 8; ++jt) {
        v16bf bb = load_b(kb, 192, j0 + jt * 16, 0, lane);
        v8f z = {};
        acc[jt] = WMMA_BF16(a, bb, z);
    }

    // Phase 2a: row max (lane -> 16-lane half -> across waves via LDS)
    float rmax[8];
    #pragma unroll
    for (int r = 0; r < 8; ++r) {
        float m = acc[0][r];
        #pragma unroll
        for (int jt = 1; jt < 8; ++jt) m = fmaxf(m, acc[jt][r]);
        for (int o = 8; o > 0; o >>= 1) m = fmaxf(m, __shfl_xor(m, o, 32));
        rmax[r] = m;
    }
    if (ln == 0) {
        #pragma unroll
        for (int r = 0; r < 8; ++r) red[wave * 16 + hf * 8 + r] = rmax[r];
    }
    __syncthreads();
    #pragma unroll
    for (int r = 0; r < 8; ++r) {
        float m = red[hf * 8 + r];
        #pragma unroll
        for (int w = 1; w < 8; ++w) m = fmaxf(m, red[w * 16 + hf * 8 + r]);
        rmax[r] = m;
    }
    __syncthreads();

    // Phase 2b: exp + row sum
    float rsum[8];
    #pragma unroll
    for (int r = 0; r < 8; ++r) {
        float s = 0.0f;
        #pragma unroll
        for (int jt = 0; jt < 8; ++jt) {
            float e = __expf(acc[jt][r] - rmax[r]);
            acc[jt][r] = e; s += e;
        }
        for (int o = 8; o > 0; o >>= 1) s += __shfl_xor(s, o, 32);
        rsum[r] = s;
    }
    if (ln == 0) {
        #pragma unroll
        for (int r = 0; r < 8; ++r) red[wave * 16 + hf * 8 + r] = rsum[r];
    }
    // zero ctx tile while waiting (512 entries, 256 threads)
    ctile[threadIdx.x] = 0.0f; ctile[threadIdx.x + 256] = 0.0f;
    __syncthreads();
    #pragma unroll
    for (int r = 0; r < 8; ++r) {
        float s = 0.0f;
        #pragma unroll
        for (int w = 0; w < 8; ++w) s += red[w * 16 + hf * 8 + r];
        rsum[r] = 1.0f / s;
    }

    // Phase 3: normalize; single NT write of attn to d_out; stash bf16 in LDS
    float* arow = attn + (size_t)bh * SS * SS + (size_t)i0 * SS + j0;
    #pragma unroll
    for (int r = 0; r < 8; ++r) {
        int row = hf * 8 + r;
        #pragma unroll
        for (int jt = 0; jt < 8; ++jt) {
            float p = acc[jt][r] * rsum[r];
            __builtin_nontemporal_store(p, &arow[(size_t)row * SS + jt * 16 + ln]);
            Psh[row * LDP + j0 + jt * 16 + ln] = (__bf16)p;
        }
    }
    __syncthreads();
    // make the async-staged V slice visible to this wave's ds reads
    asm volatile("s_wait_asynccnt 0x0" ::: "memory");

    // Phase 4: ctx partials: rows i0..+15 x d 0..31, wave handles K in [j0, j0+128)
    const __bf16* Bv = Vsh + wave * 32 * VROW;     // rows d=0..31, local cols 0..127
    v8f c0 = {}, c1 = {};
    #pragma unroll
    for (int kk = 0; kk < 128; kk += 32) {
        const __bf16* pA = Psh + ln * LDP + j0 + kk + hf * 8;
        V16U af; af.h[0] = *(const v8bf*)(pA); af.h[1] = *(const v8bf*)(pA + 16);
        int koff = hf << 3;
        const __bf16* pb0 = Bv + (ln)      * VROW + kk + koff;
        const __bf16* pb1 = Bv + (16 + ln) * VROW + kk + koff;
        V16U b0u, b1u;
        b0u.h[0] = *(const v8bf*)(pb0); b0u.h[1] = *(const v8bf*)(pb0 + 16);
        b1u.h[0] = *(const v8bf*)(pb1); b1u.h[1] = *(const v8bf*)(pb1 + 16);
        v16bf b0 = b0u.v, b1 = b1u.v;
        c0 = WMMA_BF16(af.v, b0, c0);
        c1 = WMMA_BF16(af.v, b1, c1);
    }
    #pragma unroll
    for (int r = 0; r < 8; ++r) {
        int row = hf * 8 + r;
        atomicAdd(&ctile[row * 32 + ln], c0[r]);
        atomicAdd(&ctile[row * 32 + 16 + ln], c1[r]);
    }
    __syncthreads();
    #pragma unroll
    for (int q = 0; q < 2; ++q) {
        int idx = threadIdx.x + q * 256;
        int row = idx >> 5, d = idx & 31;
        ctx[((size_t)b * SS + i0 + row) * DD + hh * DH + d] = ctile[idx];
    }
}

// ---- y = ctx @ W_o + b_o + h (pre-LN residual sum) ----
__global__ void __launch_bounds__(256) k_out_proj(const float* __restrict__ ctx, const __bf16* __restrict__ Wt,
                           const float* __restrict__ bias, const float* __restrict__ hres,
                           float* __restrict__ y) {
    int wave = threadIdx.x >> 5, lane = threadIdx.x & 31;
    int tm = blockIdx.x * 8 + wave;
    v8f acc[4] = {};
    for (int k0 = 0; k0 < DD; k0 += 32) {
        v16bf a = load_a_f32(ctx, DD, tm * 16, k0, DD, lane);
        #pragma unroll
        for (int nt = 0; nt < 4; ++nt)
            acc[nt] = WMMA_BF16(a, load_b(Wt, DD, nt * 16, k0, lane), acc[nt]);
    }
    int rbase = tm * 16 + ((lane >> 4) << 3);
    #pragma unroll
    for (int nt = 0; nt < 4; ++nt) {
        int n = nt * 16 + (lane & 15);
        float bn = bias[n];
        #pragma unroll
        for (int r = 0; r < 8; ++r) {
            size_t i = (size_t)(rbase + r) * DD + n;
            y[i] = acc[nt][r] + bn + hres[i];
        }
    }
}

// ---- LayerNorm over 64-wide rows (wave per row); optional bf16 copy ----
__global__ void k_ln(const float* __restrict__ y, const float* __restrict__ g,
                     const float* __restrict__ be, float* __restrict__ out, __bf16* outb) {
    int lane = threadIdx.x & 31;
    size_t row = (size_t)blockIdx.x * 8 + (threadIdx.x >> 5);
    const float* p = y + row * DD;
    float a = p[lane], b = p[lane + 32];
    float s = a + b;
    for (int o = 16; o > 0; o >>= 1) s += __shfl_xor(s, o, 32);
    float mu = s * (1.0f / 64.0f);
    float da = a - mu, db = b - mu;
    float vs = da * da + db * db;
    for (int o = 16; o > 0; o >>= 1) vs += __shfl_xor(vs, o, 32);
    float rstd = rsqrtf(vs * (1.0f / 64.0f) + 1e-5f);
    float o1 = da * rstd * g[lane] + be[lane];
    float o2 = db * rstd * g[lane + 32] + be[lane + 32];
    out[row * DD + lane] = o1; out[row * DD + lane + 32] = o2;
    if (outb) { outb[row * DD + lane] = (__bf16)o1; outb[row * DD + lane + 32] = (__bf16)o2; }
}

// ---- ff1 = relu(h1 @ W1 + b1) bf16 ----
__global__ void __launch_bounds__(256) k_ff1(const __bf16* __restrict__ h1b, const __bf16* __restrict__ Wt,
                      const float* __restrict__ bias, __bf16* __restrict__ ff1) {
    int wave = threadIdx.x >> 5, lane = threadIdx.x & 31;
    int t = blockIdx.x * 8 + wave;
    int tm = t >> 2, nbase = (t & 3) * 64;
    v8f acc[4] = {};
    for (int k0 = 0; k0 < DD; k0 += 32) {
        v16bf a = load_a_bf16(h1b, DD, tm * 16, k0, lane);
        #pragma unroll
        for (int nt = 0; nt < 4; ++nt)
            acc[nt] = WMMA_BF16(a, load_b(Wt, DD, nbase + nt * 16, k0, lane), acc[nt]);
    }
    int rbase = tm * 16 + ((lane >> 4) << 3);
    #pragma unroll
    for (int nt = 0; nt < 4; ++nt) {
        int n = nbase + nt * 16 + (lane & 15);
        float bn = bias[n];
        #pragma unroll
        for (int r = 0; r < 8; ++r)
            ff1[(size_t)(rbase + r) * DFF + n] = (__bf16)fmaxf(acc[nt][r] + bn, 0.0f);
    }
}

// ---- y = ff1 @ W2 + b2 + h1 ----
__global__ void __launch_bounds__(256) k_ff2(const __bf16* __restrict__ ff1, const __bf16* __restrict__ Wt,
                      const float* __restrict__ bias, const float* __restrict__ h1,
                      float* __restrict__ y) {
    int wave = threadIdx.x >> 5, lane = threadIdx.x & 31;
    int tm = blockIdx.x * 8 + wave;
    v8f acc[4] = {};
    for (int k0 = 0; k0 < DFF; k0 += 32) {
        v16bf a = load_a_bf16(ff1, DFF, tm * 16, k0, lane);
        #pragma unroll
        for (int nt = 0; nt < 4; ++nt)
            acc[nt] = WMMA_BF16(a, load_b(Wt, DFF, nt * 16, k0, lane), acc[nt]);
    }
    int rbase = tm * 16 + ((lane >> 4) << 3);
    #pragma unroll
    for (int nt = 0; nt < 4; ++nt) {
        int n = nt * 16 + (lane & 15);
        float bn = bias[n];
        #pragma unroll
        for (int r = 0; r < 8; ++r) {
            size_t i = (size_t)(rbase + r) * DD + n;
            y[i] = acc[nt][r] + bn + h1[i];
        }
    }
}

// ---- mean pool over S ----
__global__ void k_pool(const float* __restrict__ h2, float* __restrict__ pooled) {
    int idx = blockIdx.x * blockDim.x + threadIdx.x;
    if (idx >= BB * DD) return;
    int b = idx >> 6, d = idx & 63;
    const float* p = h2 + (size_t)b * SS * DD + d;
    float s = 0.0f;
    for (int t = 0; t < SS; ++t) s += p[(size_t)t * DD];
    pooled[idx] = s * (1.0f / (float)SS);
}

// ---- classifier head ----
__global__ void k_head(const float* __restrict__ pooled, const float* __restrict__ Wc1,
                       const float* __restrict__ bc1, const float* __restrict__ Wc2,
                       const float* __restrict__ bc2, float* __restrict__ logits) {
    int i = threadIdx.x;
    if (i >= BB) return;
    float t[DD];
    for (int j = 0; j < DD; ++j) {
        float s = bc1[j];
        for (int k = 0; k < DD; ++k) s += pooled[i * DD + k] * Wc1[k * DD + j];
        t[j] = fmaxf(s, 0.0f);
    }
    for (int c = 0; c < 2; ++c) {
        float s = bc2[c];
        for (int j = 0; j < DD; ++j) s += t[j] * Wc2[j * 2 + c];
        logits[i * 2 + c] = s;
    }
}

extern "C" void kernel_launch(void* const* d_in, const int* in_sizes, int n_in,
                              void* d_out, int out_size, void* d_ws, size_t ws_size,
                              hipStream_t stream) {
    const float* x    = (const float*)d_in[0];
    const float* W_in = (const float*)d_in[1];
    const float* b_in = (const float*)d_in[2];
    const float* Wqkv = (const float*)d_in[3];
    const float* bqkv = (const float*)d_in[4];
    const float* W_o  = (const float*)d_in[5];
    const float* b_o  = (const float*)d_in[6];
    const float* g1   = (const float*)d_in[7];
    const float* be1  = (const float*)d_in[8];
    const float* W1   = (const float*)d_in[9];
    const float* b1   = (const float*)d_in[10];
    const float* W2   = (const float*)d_in[11];
    const float* b2   = (const float*)d_in[12];
    const float* g2   = (const float*)d_in[13];
    const float* be2  = (const float*)d_in[14];
    const float* Wc1  = (const float*)d_in[15];
    const float* bc1  = (const float*)d_in[16];
    const float* Wc2  = (const float*)d_in[17];
    const float* bc2  = (const float*)d_in[18];

    float* logits = (float*)d_out;
    float* attn   = logits + BB * 2;

    char* ws = (char*)d_ws;
    size_t off = 0;
    auto take = [&](size_t bytes) -> char* {
        char* p = ws + off;
        off = (off + bytes + 255) & ~(size_t)255;
        return p;
    };
    float*  h    = (float*) take((size_t)MM * DD * 4);
    __bf16* hb   = (__bf16*)take((size_t)MM * DD * 2);
    __bf16* qkv  = (__bf16*)take((size_t)MM * 192 * 2);
    __bf16* vt   = (__bf16*)take((size_t)BB * HHN * DH * SS * 2);
    float*  ctx  = (float*) take((size_t)MM * DD * 4);
    float*  y    = (float*) take((size_t)MM * DD * 4);
    float*  h1   = (float*) take((size_t)MM * DD * 4);
    __bf16* h1b  = (__bf16*)take((size_t)MM * DD * 2);
    __bf16* ff1  = (__bf16*)take((size_t)MM * DFF * 2);
    float*  pooled = (float*)take((size_t)BB * DD * 4);
    __bf16* WtIn = (__bf16*)take((size_t)DD * 160 * 2);
    __bf16* WtQ  = (__bf16*)take((size_t)192 * DD * 2);
    __bf16* WtO  = (__bf16*)take((size_t)DD * DD * 2);
    __bf16* Wt1  = (__bf16*)take((size_t)DFF * DD * 2);
    __bf16* Wt2  = (__bf16*)take((size_t)DD * DFF * 2);
    float*  h2   = h;   // h is dead after its use as residual in k_out_proj

    // weight prep
    k_tcvt<<<(DD * 160 + 255) / 256, 256, 0, stream>>>(W_in, WtIn, FFK, DD, 160);
    k_tcvt<<<(192 * DD + 255) / 256, 256, 0, stream>>>(Wqkv, WtQ, DD, 192, DD);
    k_tcvt<<<(DD * DD + 255) / 256, 256, 0, stream>>>(W_o, WtO, DD, DD, DD);
    k_tcvt<<<(DFF * DD + 255) / 256, 256, 0, stream>>>(W1, Wt1, DD, DFF, DD);
    k_tcvt<<<(DD * DFF + 255) / 256, 256, 0, stream>>>(W2, Wt2, DFF, DD, DFF);

    // input projection + PE
    k_in_proj<<<(MM / 16) / 8, 256, 0, stream>>>(x, WtIn, b_in, h, hb);
    // qkv (q pre-scaled by 1/sqrt(DH))
    k_qkv<<<(MM / 16) * 3 / 8, 256, 0, stream>>>(hb, WtQ, bqkv, qkv);
    // V transpose for ctx GEMM
    k_vt<<<(BB * HHN * DH * SS + 255) / 256, 256, 0, stream>>>(qkv, vt);
    // fused attention: scores + softmax + attn write + ctx
    k_attn_fused<<<dim3(SS / 16, BB * HHN), 256, 0, stream>>>(qkv, vt, attn, ctx);
    // output projection + residual, LN1
    k_out_proj<<<(MM / 16) / 8, 256, 0, stream>>>(ctx, WtO, b_o, h, y);
    k_ln<<<MM / 8, 256, 0, stream>>>(y, g1, be1, h1, h1b);
    // FFN + residual, LN2
    k_ff1<<<(MM / 16) * 4 / 8, 256, 0, stream>>>(h1b, Wt1, b1, ff1);
    k_ff2<<<(MM / 16) / 8, 256, 0, stream>>>(ff1, Wt2, b2, h1, y);
    k_ln<<<MM / 8, 256, 0, stream>>>(y, g2, be2, h2, (__bf16*)nullptr);
    // pool + head
    k_pool<<<(BB * DD + 255) / 256, 256, 0, stream>>>(h2, pooled);
    k_head<<<1, 64, 0, stream>>>(pooled, Wc1, bc1, Wc2, bc2, logits);
}
